// InformerEncoder_17910013624422
// MI455X (gfx1250) — matrix operand order repaired and checked by reading, not compiled, not run on hardware
//
#include <hip/hip_runtime.h>
#include <hip/hip_bf16.h>
#include <math.h>

// ---------------- constants ----------------
#define BSZ 4
#define SEQ 1024
#define DM 512
#define NH 8
#define DK 64
#define DFF 2048
#define NDEPTH 2
#define NU 35            // u = top-k count = 5*ceil(ln(1024))
#define NSAMP 35         // sample_k
#define ROWS (BSZ*SEQ)   // 4096
#define X_ELEMS ((size_t)ROWS*DM)            // 2,097,152
#define ATT_ELEMS ((size_t)BSZ*NH*SEQ*SEQ)   // 33,554,432

typedef __bf16 bf16;
typedef bf16 v16bf __attribute__((ext_vector_type(16)));
typedef bf16 v8bf  __attribute__((ext_vector_type(8)));
typedef float v8f  __attribute__((ext_vector_type(8)));

__device__ __forceinline__ unsigned short bfbits(float f) {
  union { float f; unsigned u; } c; c.f = f;
  unsigned r = c.u + 0x7FFFu + ((c.u >> 16) & 1u);   // round-to-nearest-even
  return (unsigned short)(r >> 16);
}
__device__ __forceinline__ unsigned pack2bf(float x, float y) {
  return (unsigned)bfbits(x) | ((unsigned)bfbits(y) << 16);
}

__device__ __forceinline__ unsigned hash_u32(unsigned x) {
  x ^= x >> 16; x *= 0x7feb352dU;
  x ^= x >> 15; x *= 0x846ca68bU;
  x ^= x >> 16; return x;
}

// ---------------- WMMA bf16 GEMM: C[M,N] = A[M,K]*W[K,N] + bias, opt ReLU ----
// block = 256 threads (8 waves), tile 128x128, K step 32, double-buffered LDS.
// LDS holds bf16 pairs packed as u32: lds_a[buf][m][kpair], lds_b[buf][n][kpair].
__global__ __launch_bounds__(256)
void gemm_bf16_kernel(const float* __restrict__ A, const float* __restrict__ W,
                      const float* __restrict__ bias, float* __restrict__ C,
                      int M, int N, int K, int relu) {
  __shared__ unsigned lds_a[2][128][16];
  __shared__ unsigned lds_b[2][128][16];
  const int tid  = threadIdx.x;
  const int lane = tid & 31;
  const int wid  = tid >> 5;
  const int wm   = wid >> 2;        // 0..1  -> 64 rows each
  const int wn   = wid & 3;         // 0..3  -> 32 cols each
  const int half = lane >> 4;       // wave32 half
  const int lrow = lane & 15;
  const int m0 = blockIdx.y * 128;
  const int n0 = blockIdx.x * 128;

  v8f acc[4][2] = {};
  unsigned ra[8], rb[8];            // staged (packed) tile for next LDS fill

  // fetch one K-tile (k0) from global into registers, converting f32->bf16x2
  auto fetch = [&](int k0) {
    // A tile: 128 rows x 8 float4 = 1024 chunks, 4 per thread
#pragma unroll
    for (int it = 0; it < 4; ++it) {
      int p = tid + it * 256;
      int row = p >> 3, kq = p & 7;
      const float4 f = *(const float4*)(A + (size_t)(m0 + row) * K + k0 + kq * 4);
      ra[2 * it]     = pack2bf(f.x, f.y);
      ra[2 * it + 1] = pack2bf(f.z, f.w);
    }
    // B tile (32x128) transposed into [n][kpair]: 2048 pairs, 8 per thread
#pragma unroll
    for (int it = 0; it < 8; ++it) {
      int p = tid + it * 256;
      int kp = p >> 7, n = p & 127;
      const float* src = W + (size_t)(k0 + kp * 2) * N + n0 + n;
      rb[it] = pack2bf(src[0], src[N]);
    }
  };

  fetch(0);
  int buf = 0;
  for (int k0 = 0; k0 < K; k0 += 32) {
    // commit staged registers to LDS buffer `buf`
#pragma unroll
    for (int it = 0; it < 4; ++it) {
      int p = tid + it * 256;
      int row = p >> 3, kq = p & 7;
      lds_a[buf][row][kq * 2]     = ra[2 * it];
      lds_a[buf][row][kq * 2 + 1] = ra[2 * it + 1];
    }
#pragma unroll
    for (int it = 0; it < 8; ++it) {
      int p = tid + it * 256;
      int kp = p >> 7, n = p & 127;
      lds_b[buf][n][kp] = rb[it];
    }
    __syncthreads();

    // overlap: fetch next K-tile while WMMAs consume current buffer
    if (k0 + 32 < K) fetch(k0 + 32);

    // ---- fragments (16-bit A 16x32 / B 32x16 layouts) + WMMA ----
    v16bf afrag[4], bfrag[2];
#pragma unroll
    for (int tm = 0; tm < 4; ++tm) {
      int r = wm * 64 + tm * 16 + lrow;
      v8bf lo = *(const v8bf*)&lds_a[buf][r][half * 4];      // K = half*8 .. +7
      v8bf hi = *(const v8bf*)&lds_a[buf][r][8 + half * 4];  // K = 16+half*8 .. +7
      afrag[tm] = __builtin_shufflevector(lo, hi, 0,1,2,3,4,5,6,7,8,9,10,11,12,13,14,15);
    }
#pragma unroll
    for (int tn = 0; tn < 2; ++tn) {
      int cn = wn * 32 + tn * 16 + lrow;
      v8bf lo = *(const v8bf*)&lds_b[buf][cn][half * 4];     // K = half*8 .. +7
      v8bf hi = *(const v8bf*)&lds_b[buf][cn][8 + half * 4]; // K = 16+half*8 .. +7
      bfrag[tn] = __builtin_shufflevector(lo, hi, 0,1,2,3,4,5,6,7,8,9,10,11,12,13,14,15);
    }
#pragma unroll
    for (int tm = 0; tm < 4; ++tm)
#pragma unroll
      for (int tn = 0; tn < 2; ++tn)
        acc[tm][tn] = __builtin_amdgcn_wmma_f32_16x16x32_bf16(
            false, afrag[tm], false, bfrag[tn], (short)0, acc[tm][tn], false, false);

    buf ^= 1;   // next store targets the other buffer -> one barrier/iteration
  }

  // ---- epilogue: bias (+ReLU), f32 C/D layout: lane->N, vgpr(+8*half)->M ----
#pragma unroll
  for (int tm = 0; tm < 4; ++tm) {
#pragma unroll
    for (int tn = 0; tn < 2; ++tn) {
      int col = n0 + wn * 32 + tn * 16 + lrow;
      float bv = bias[col];
#pragma unroll
      for (int v = 0; v < 8; ++v) {
        int row = m0 + wm * 64 + tm * 16 + v + half * 8;
        float val = acc[tm][tn][v] + bv;
        if (relu) val = fmaxf(val, 0.0f);
        C[(size_t)row * N + col] = val;
      }
    }
  }
}

// ---------------- PE + LayerNorm (input stage) ----------------
__global__ __launch_bounds__(256)
void pe_ln_kernel(const float* __restrict__ x, const float* __restrict__ g,
                  const float* __restrict__ be, float* __restrict__ out) {
  __shared__ float buf[DM];
  __shared__ float red[256];
  const int row = blockIdx.x;           // b*SEQ + l
  const int l   = row & (SEQ - 1);
  const int tid = threadIdx.x;
  const size_t base = (size_t)row * DM;
  for (int c = tid; c < DM; c += 256) {
    int i2 = c & ~1;                    // exponent index value (0,2,4,...)
    float ang = (float)l * __expf(-(float)i2 * (9.210340371976184f / (float)DM));
    float pe = (c & 1) ? __cosf(ang) : __sinf(ang);
    buf[c] = x[base + c] + pe;
  }
  __syncthreads();
  red[tid] = buf[tid] + buf[tid + 256];
  __syncthreads();
  for (int s = 128; s > 0; s >>= 1) { if (tid < s) red[tid] += red[tid + s]; __syncthreads(); }
  float mu = red[0] * (1.0f / DM);
  __syncthreads();
  float d0 = buf[tid] - mu, d1 = buf[tid + 256] - mu;
  red[tid] = d0 * d0 + d1 * d1;
  __syncthreads();
  for (int s = 128; s > 0; s >>= 1) { if (tid < s) red[tid] += red[tid + s]; __syncthreads(); }
  float rs = rsqrtf(red[0] * (1.0f / DM) + 1e-5f);
  for (int c = tid; c < DM; c += 256)
    out[base + c] = (buf[c] - mu) * rs * g[c] + be[c];
}

// ---------------- residual add + LayerNorm ----------------
__global__ __launch_bounds__(256)
void addln_kernel(const float* __restrict__ x, const float* __restrict__ y,
                  const float* __restrict__ g, const float* __restrict__ be,
                  float* __restrict__ out) {
  __shared__ float buf[DM];
  __shared__ float red[256];
  const int row = blockIdx.x;
  const int tid = threadIdx.x;
  const size_t base = (size_t)row * DM;
  for (int c = tid; c < DM; c += 256) buf[c] = x[base + c] + y[base + c];
  __syncthreads();
  red[tid] = buf[tid] + buf[tid + 256];
  __syncthreads();
  for (int s = 128; s > 0; s >>= 1) { if (tid < s) red[tid] += red[tid + s]; __syncthreads(); }
  float mu = red[0] * (1.0f / DM);
  __syncthreads();
  float d0 = buf[tid] - mu, d1 = buf[tid + 256] - mu;
  red[tid] = d0 * d0 + d1 * d1;
  __syncthreads();
  for (int s = 128; s > 0; s >>= 1) { if (tid < s) red[tid] += red[tid + s]; __syncthreads(); }
  float rs = rsqrtf(red[0] * (1.0f / DM) + 1e-5f);
  for (int c = tid; c < DM; c += 256)
    out[base + c] = (buf[c] - mu) * rs * g[c] + be[c];
}

// ---------------- ProbSparse sparsity measure M[b,h,l] ----------------
__global__ __launch_bounds__(256)
void probm_kernel(const float* __restrict__ q, const float* __restrict__ k,
                  float* __restrict__ M, int layer) {
  int idx = blockIdx.x * blockDim.x + threadIdx.x;     // (b*NH+h)*SEQ + l
  if (idx >= BSZ * NH * SEQ) return;
  int l  = idx & (SEQ - 1);
  int bh = idx >> 10;
  int h  = bh & (NH - 1);
  int b  = bh >> 3;
  const float* qrow = q + ((size_t)(b * SEQ + l)) * DM + h * DK;
  float mx = -1e30f, sm = 0.0f;
  for (int s = 0; s < NSAMP; ++s) {
    unsigned hv = hash_u32((unsigned)(((layer * SEQ + l) * 64) + s) * 0x9E3779B9u + 0x85EBCA6Bu);
    int ki = (int)(hv & (SEQ - 1));
    const float* kr = k + ((size_t)(b * SEQ + ki)) * DM + h * DK;
    float dp = 0.0f;
#pragma unroll 8
    for (int d = 0; d < DK; ++d) dp += qrow[d] * kr[d];
    mx = fmaxf(mx, dp);
    sm += dp;
  }
  M[idx] = mx - sm * (1.0f / (float)SEQ);
}

// ---------------- iterative top-NU per (b,h) ----------------
__global__ __launch_bounds__(256)
void topk_kernel(const float* __restrict__ M, int* __restrict__ mtop) {
  __shared__ float sM[SEQ];
  __shared__ float rv[256];
  __shared__ int   ri[256];
  const int bh = blockIdx.x;
  const int tid = threadIdx.x;
  for (int l = tid; l < SEQ; l += 256) sM[l] = M[(size_t)bh * SEQ + l];
  __syncthreads();
  for (int it = 0; it < NU; ++it) {
    float bvv = -1e30f; int bii = 0;
    for (int l = tid; l < SEQ; l += 256)
      if (sM[l] > bvv) { bvv = sM[l]; bii = l; }
    rv[tid] = bvv; ri[tid] = bii;
    __syncthreads();
    for (int s = 128; s > 0; s >>= 1) {
      if (tid < s && rv[tid + s] > rv[tid]) { rv[tid] = rv[tid + s]; ri[tid] = ri[tid + s]; }
      __syncthreads();
    }
    if (tid == 0) { mtop[bh * NU + it] = ri[0]; sM[ri[0]] = -1e30f; }
    __syncthreads();
  }
}

// ---------------- V mean over L per (b,h) ----------------
__global__ __launch_bounds__(64)
void vmean_kernel(const float* __restrict__ v, float* __restrict__ vmean) {
  const int bh = blockIdx.x;
  const int b = bh >> 3, h = bh & (NH - 1);
  const int dd = threadIdx.x;
  float s = 0.0f;
  for (int l = 0; l < SEQ; ++l)
    s += v[((size_t)(b * SEQ + l)) * DM + h * DK + dd];
  vmean[bh * DK + dd] = s * (1.0f / (float)SEQ);
}

// ---------------- broadcast V-mean into ctx (B,L,H*DK) ----------------
__global__ __launch_bounds__(256)
void ctxfill_kernel(const float* __restrict__ vmean, float* __restrict__ ctx) {
  size_t idx = (size_t)blockIdx.x * blockDim.x + threadIdx.x;
  if (idx >= X_ELEMS) return;
  int c = (int)(idx & (DM - 1));
  int b = (int)(idx >> 19);               // idx / (SEQ*DM)
  int h = c >> 6, dd = c & (DK - 1);
  ctx[idx] = vmean[(b * NH + h) * DK + dd];
}

// ---------------- float4 fill (attn base = 1/L) ----------------
__global__ __launch_bounds__(256)
void fill_kernel(float* __restrict__ p, float val, size_t n4) {
  size_t i = (size_t)blockIdx.x * blockDim.x + threadIdx.x;
  size_t stride = (size_t)gridDim.x * blockDim.x;
  float4 v = make_float4(val, val, val, val);
  for (; i < n4; i += stride) ((float4*)p)[i] = v;
}

// ------ reduced attention: scores row + softmax + ctx/attn scatter ------
__global__ __launch_bounds__(256)
void attn_kernel(const float* __restrict__ q, const float* __restrict__ k,
                 const float* __restrict__ v, const int* __restrict__ mtop,
                 float* __restrict__ ctx, float* __restrict__ attns) {
  __shared__ float qs[DK];
  __shared__ float ssc[SEQ];
  __shared__ float red[256];
  const int blk = blockIdx.x;             // (b*NH+h)*NU + i
  const int i  = blk % NU;
  const int bh = blk / NU;
  const int h = bh & (NH - 1), b = bh >> 3;
  const int tid = threadIdx.x;
  const int qi = mtop[bh * NU + i];

  if (tid < DK) qs[tid] = q[((size_t)(b * SEQ + qi)) * DM + h * DK + tid];
  __syncthreads();

  float lmax = -1e30f;
  for (int l = tid; l < SEQ; l += 256) {
    const float* kr = k + ((size_t)(b * SEQ + l)) * DM + h * DK;
    float dp = 0.0f;
#pragma unroll 8
    for (int d = 0; d < DK; ++d) dp += qs[d] * kr[d];
    dp *= 0.125f;                         // 1/sqrt(64)
    ssc[l] = dp;
    lmax = fmaxf(lmax, dp);
  }
  red[tid] = lmax; __syncthreads();
  for (int s = 128; s > 0; s >>= 1) { if (tid < s) red[tid] = fmaxf(red[tid], red[tid + s]); __syncthreads(); }
  float m = red[0];
  __syncthreads();
  float lsum = 0.0f;
  for (int l = tid; l < SEQ; l += 256) { float e = __expf(ssc[l] - m); ssc[l] = e; lsum += e; }
  red[tid] = lsum; __syncthreads();
  for (int s = 128; s > 0; s >>= 1) { if (tid < s) red[tid] += red[tid + s]; __syncthreads(); }
  float inv = 1.0f / red[0];

  float* arow = attns + ((size_t)(bh * SEQ + qi)) * SEQ;
  for (int l = tid; l < SEQ; l += 256) arow[l] = ssc[l] * inv;

  if (tid < DK) {
    float acc = 0.0f;
    for (int l = 0; l < SEQ; ++l)
      acc += ssc[l] * v[((size_t)(b * SEQ + l)) * DM + h * DK + tid];
    ctx[((size_t)(b * SEQ + qi)) * DM + h * DK + tid] = acc * inv;
  }
}

// ---------------- host side ----------------
extern "C" void kernel_launch(void* const* d_in, const int* in_sizes, int n_in,
                              void* d_out, int out_size, void* d_ws, size_t ws_size,
                              hipStream_t stream) {
  const float* x_in = (const float*)d_in[0];
  const float* Wq = (const float*)d_in[1];  const float* bq  = (const float*)d_in[2];
  const float* Wk = (const float*)d_in[3];  const float* bk  = (const float*)d_in[4];
  const float* Wv = (const float*)d_in[5];  const float* bv  = (const float*)d_in[6];
  const float* Wo = (const float*)d_in[7];  const float* bo  = (const float*)d_in[8];
  const float* W1 = (const float*)d_in[9];  const float* b1f = (const float*)d_in[10];
  const float* W2 = (const float*)d_in[11]; const float* b2f = (const float*)d_in[12];
  const float* g1 = (const float*)d_in[13]; const float* be1 = (const float*)d_in[14];
  const float* g2 = (const float*)d_in[15]; const float* be2 = (const float*)d_in[16];
  const float* g0 = (const float*)d_in[17]; const float* be0 = (const float*)d_in[18];

  float* ws = (float*)d_ws;
  float* xbuf  = ws;                    // ROWS x DM
  float* qb    = xbuf  + X_ELEMS;
  float* kb    = qb    + X_ELEMS;
  float* vb    = kb    + X_ELEMS;
  float* ctx   = vb    + X_ELEMS;
  float* yb    = ctx   + X_ELEMS;
  float* ff    = yb    + X_ELEMS;       // ROWS x DFF
  float* Mb    = ff    + (size_t)ROWS * DFF;
  float* vmean = Mb    + (size_t)BSZ * NH * SEQ;
  int*   mtop  = (int*)(vmean + BSZ * NH * DK);

  float* out   = (float*)d_out;
  float* xout  = out;                               // (B,L,DM)
  float* attnL[NDEPTH];
  attnL[0] = out + X_ELEMS;
  attnL[1] = attnL[0] + ATT_ELEMS;

  // x = LN(x + PE)
  pe_ln_kernel<<<ROWS, 256, 0, stream>>>(x_in, g0, be0, xbuf);

  for (int d = 0; d < NDEPTH; ++d) {
    const float* Wq_d = Wq + (size_t)d * DM * DM;  const float* bq_d = bq + (size_t)d * DM;
    const float* Wk_d = Wk + (size_t)d * DM * DM;  const float* bk_d = bk + (size_t)d * DM;
    const float* Wv_d = Wv + (size_t)d * DM * DM;  const float* bv_d = bv + (size_t)d * DM;
    const float* Wo_d = Wo + (size_t)d * DM * DM;  const float* bo_d = bo + (size_t)d * DM;
    const float* W1_d = W1 + (size_t)d * DM * DFF; const float* b1_d = b1f + (size_t)d * DFF;
    const float* W2_d = W2 + (size_t)d * DFF * DM; const float* b2_d = b2f + (size_t)d * DM;

    dim3 gProj(DM / 128, ROWS / 128);   // 4 x 32
    gemm_bf16_kernel<<<gProj, 256, 0, stream>>>(xbuf, Wq_d, bq_d, qb, ROWS, DM, DM, 0);
    gemm_bf16_kernel<<<gProj, 256, 0, stream>>>(xbuf, Wk_d, bk_d, kb, ROWS, DM, DM, 0);
    gemm_bf16_kernel<<<gProj, 256, 0, stream>>>(xbuf, Wv_d, bv_d, vb, ROWS, DM, DM, 0);

    probm_kernel<<<(BSZ * NH * SEQ + 255) / 256, 256, 0, stream>>>(qb, kb, Mb, d);
    topk_kernel<<<BSZ * NH, 256, 0, stream>>>(Mb, mtop);
    vmean_kernel<<<BSZ * NH, 64, 0, stream>>>(vb, vmean);
    ctxfill_kernel<<<(unsigned)(X_ELEMS / 256), 256, 0, stream>>>(vmean, ctx);
    fill_kernel<<<2048, 256, 0, stream>>>(attnL[d], 1.0f / (float)SEQ, ATT_ELEMS / 4);
    attn_kernel<<<BSZ * NH * NU, 256, 0, stream>>>(qb, kb, vb, mtop, ctx, attnL[d]);

    gemm_bf16_kernel<<<gProj, 256, 0, stream>>>(ctx, Wo_d, bo_d, yb, ROWS, DM, DM, 0);
    addln_kernel<<<ROWS, 256, 0, stream>>>(xbuf, yb, g1 + (size_t)d * DM, be1 + (size_t)d * DM, xbuf);

    dim3 gFF1(DFF / 128, ROWS / 128);   // 16 x 32
    gemm_bf16_kernel<<<gFF1, 256, 0, stream>>>(xbuf, W1_d, b1_d, ff, ROWS, DFF, DM, 1);
    gemm_bf16_kernel<<<gProj, 256, 0, stream>>>(ff, W2_d, b2_d, yb, ROWS, DM, DFF, 0);

    float* lnout = (d == NDEPTH - 1) ? xout : xbuf;
    addln_kernel<<<ROWS, 256, 0, stream>>>(xbuf, yb, g2 + (size_t)d * DM, be2 + (size_t)d * DM, lnout);
  }
}
